// GCRec_80693845557944
// MI455X (gfx1250) — matrix operand order repaired and checked by reading, not compiled, not run on hardware
//
#include <hip/hip_runtime.h>
#include <hip/hip_bf16.h>
#include <stdint.h>

// ---------------- problem constants (match reference) ----------------
#define U_N      60000
#define I_N      30000
#define D_DIM    64
#define L_LAY    3
#define E_EDGES  1000000
#define EU_EDGES 600000
#define EI_EDGES 300000
#define B_BATCH  4096
#define N_NODES  (U_N + I_N)
#define TEMP_T   0.2f
#define REG_W    0.0001f
#define SSL_W    0.1f
#define INTRA_W  0.1f

typedef __attribute__((ext_vector_type(16))) _Float16 v16h;
typedef __attribute__((ext_vector_type(8)))  _Float16 v8h;
typedef __attribute__((ext_vector_type(8)))  float    v8f;
typedef __attribute__((ext_vector_type(4)))  unsigned int v4u;
typedef __attribute__((ext_vector_type(8)))  int      v8i;
typedef __attribute__((ext_vector_type(4)))  int      v4i;

#if __has_builtin(__builtin_amdgcn_tensor_load_to_lds)
#define HAVE_TDM 1
#else
#define HAVE_TDM 0
#endif

// ---------------- elementwise utility kernels ----------------
__global__ void k_zero(float* p, long n) {
    long i = (long)blockIdx.x * blockDim.x + threadIdx.x;
    if (i < n) p[i] = 0.0f;
}

__global__ void k_concat(const float* __restrict__ u, const float* __restrict__ it,
                         float* __restrict__ out) {
    long i = (long)blockIdx.x * blockDim.x + threadIdx.x;
    const long total = (long)N_NODES * D_DIM;
    if (i >= total) return;
    const long usz = (long)U_N * D_DIM;
    out[i] = (i < usz) ? u[i] : it[i - usz];
}

__global__ void k_acc(float* __restrict__ acc, const float* __restrict__ x, long n) {
    long i = (long)blockIdx.x * blockDim.x + threadIdx.x;
    if (i < n) acc[i] += x[i];
}

// ---------------- graph scatter: y[row] += v * x[col], one wave per edge ----------------
__global__ void k_scatter(const int* __restrict__ rows, const int* __restrict__ cols,
                          const float* __restrict__ vals,
                          const float* __restrict__ x, float* __restrict__ y, int nnz) {
    long tid = (long)blockIdx.x * blockDim.x + threadIdx.x;
    int e    = (int)(tid >> 5);
    int lane = (int)(tid & 31);
    if (e >= nnz) return;
    int   r = rows[e];
    int   c = cols[e];
    float v = vals ? vals[e] : 1.0f;
    const float* xr = x + (long)c * D_DIM;
    float*       yr = y + (long)r * D_DIM;
    __builtin_prefetch(yr, 1, 0);   // global_prefetch of destination row
    atomicAdd(yr + lane,      v * xr[lane]);
    atomicAdd(yr + 32 + lane, v * xr[32 + lane]);
}

// ---------------- degree histograms ----------------
__global__ void k_deg(const int* __restrict__ src, const int* __restrict__ dst,
                      float* __restrict__ degO, float* __restrict__ degI, int n) {
    long i = (long)blockIdx.x * blockDim.x + threadIdx.x;
    if (i >= n) return;
    atomicAdd(&degO[src[i]], 1.0f);
    atomicAdd(&degI[dst[i]], 1.0f);
}

// ---------------- pack W (f32 64x64) into f16 WMMA-B lane order ----------------
// packed[((kt*4+jt)*32+lane)*16 + i] = (f16) W[(kt*32 + off16(lane) + i)*64 + jt*16 + lane%16]
// First 4096 halves: W ; next 4096 halves: rW.
__global__ void k_packW(const float* __restrict__ W, const float* __restrict__ rW,
                        _Float16* __restrict__ packed) {
    int t = blockIdx.x * blockDim.x + threadIdx.x;
    if (t >= 8192) return;
    int idx  = t & 4095;
    int i    = idx & 15;
    int lane = (idx >> 4) & 31;
    int jt   = (idx >> 9) & 3;
    int kt   = (idx >> 11) & 1;
    const float* src = (t >= 4096) ? rW : W;
    int off16 = (lane < 16) ? 0 : 16;
    int ncol  = jt * 16 + (lane & 15);
    packed[t] = (_Float16)src[(kt * 32 + off16 + i) * D_DIM + ncol];
}

// ---------------- WMMA tile loaders (contiguous k = 16h + s order, wave32) ----------------
// A (16x32 f16) from f32 matrix: lane<16 -> K = kt*32 + {0..15} ; lane>=16 -> +16
__device__ inline v16h load_a_from_f32(const float* __restrict__ X, int m, int kt, int lane) {
    int off16 = (lane < 16) ? 0 : 16;
    const float* p = X + (long)m * D_DIM + kt * 32 + off16;
    v16h a;
#pragma unroll
    for (int i = 0; i < 8; ++i) {
        a[i]     = (_Float16)p[i];
        a[8 + i] = (_Float16)p[8 + i];
    }
    return a;
}

// B tile from LDS-staged packed panel: two ds_load_b128 per tile
__device__ inline v16h load_b_lds(const _Float16* sW, int jt, int kt, int lane) {
    const v8h* p = (const v8h*)(sW + (((kt * 4 + jt) * 32 + lane) << 4));
    v8h lo = p[0], hi = p[1];
    v16h b;
#pragma unroll
    for (int i = 0; i < 8; ++i) { b[i] = lo[i]; b[8 + i] = hi[i]; }
    return b;
}

// ---------------- encoder GEMM: one wave computes a 16x64 tile ----------------
// mode 0: out = (X @ W) * rsqrt(max(deg,1))[row]
// mode 1: out = (X @ W) + agg[row,col]*rsqrt(max(deg,1))[row] + bvec[col] + rbvec[col]
__global__ void k_encoder_gemm(const float* __restrict__ X, const _Float16* __restrict__ packW,
                               const float* __restrict__ bvec, const float* __restrict__ rbvec,
                               const float* __restrict__ deg, const float* __restrict__ agg,
                               float* __restrict__ out, int n, int mode) {
    __shared__ __attribute__((aligned(32))) _Float16 sW[4096];   // 8 KB packed B panel

#if HAVE_TDM
    // Stage the packed 8KB B panel into LDS with the Tensor Data Mover.
    if (threadIdx.x == 0) {
        unsigned long long ga = (unsigned long long)(uintptr_t)packW;
        unsigned lds_off =
            (unsigned)(uintptr_t)(__attribute__((address_space(3))) void*)(void*)sW;
        v4u g0;
        g0[0] = 1u;                                   // count=1 (valid descriptor)
        g0[1] = lds_off;                              // lds_addr (bytes)
        g0[2] = (unsigned)(ga & 0xffffffffu);         // global_addr[31:0]
        g0[3] = (unsigned)((ga >> 32) & 0x01ffffffu)  // global_addr[56:32]
              | (2u << 30);                           // type = 2 ("image")
        v8i g1;
        g1[0] = 0x00030000;          // workgroup_mask=0, data_size=3 (8B elements)
        g1[1] = (int)(1024u << 16);  // tensor_dim0 = 1024 (8KB / 8B)
        g1[2] = (int)(1u << 16);     // tensor_dim0_hi=0, tensor_dim1 = 1
        g1[3] = (int)(1024u << 16);  // tensor_dim1_hi=0, tile_dim0 = 1024
        g1[4] = 1;                   // tile_dim1 = 1, tile_dim2 = 0
        g1[5] = 1024;                // tensor_dim0_stride = 1024
        g1[6] = 0;
        g1[7] = 0;
        v4i gz = {0, 0, 0, 0};
#if __clang_major__ >= 23
        v8i gz8 = {0, 0, 0, 0, 0, 0, 0, 0};
        __builtin_amdgcn_tensor_load_to_lds(g0, g1, gz, gz, gz8, 0);
#else
        __builtin_amdgcn_tensor_load_to_lds(g0, g1, gz, gz, 0);
#endif
#if __has_builtin(__builtin_amdgcn_s_wait_tensorcnt)
        __builtin_amdgcn_s_wait_tensorcnt(0);
#else
        asm volatile("s_wait_tensorcnt 0x0" ::: "memory");
#endif
    }
    __syncthreads();
#else
    // Fallback: cooperative copy of the packed panel into LDS.
    {
        const uint4* gsrc = (const uint4*)packW;
        uint4*       sdst = (uint4*)sW;
        for (int i = threadIdx.x; i < 512; i += blockDim.x) sdst[i] = gsrc[i];
        __syncthreads();
    }
#endif

    int lane = threadIdx.x & 31;
    int wid  = blockIdx.x * (blockDim.x >> 5) + (threadIdx.x >> 5);
    int rowBase = wid * 16;
    if (rowBase >= n) return;

    int m = rowBase + (lane & 15);
    v16h a0 = load_a_from_f32(X, m, 0, lane);
    v16h a1 = load_a_from_f32(X, m, 1, lane);

    int rhalf = (lane < 16) ? 0 : 8;
    float rscale[8];
#pragma unroll
    for (int r = 0; r < 8; ++r) {
        int row = rowBase + r + rhalf;
        rscale[r] = rsqrtf(fmaxf(deg[row], 1.0f));
    }

#pragma unroll
    for (int j = 0; j < 4; ++j) {
        v16h b0 = load_b_lds(sW, j, 0, lane);
        v16h b1 = load_b_lds(sW, j, 1, lane);
        v8f c = {};
        c = __builtin_amdgcn_wmma_f32_16x16x32_f16(false, a0, false, b0, (short)0, c, false, false);
        c = __builtin_amdgcn_wmma_f32_16x16x32_f16(false, a1, false, b1, (short)0, c, false, false);

        int col = j * 16 + (lane & 15);
        float badd = 0.0f;
        if (mode == 1) badd = bvec[col] + rbvec[col];
#pragma unroll
        for (int r = 0; r < 8; ++r) {
            long row = rowBase + r + rhalf;
            if (mode == 0) {
                out[row * D_DIM + col] = c[r] * rscale[r];
            } else {
                out[row * D_DIM + col] = c[r] + agg[row * D_DIM + col] * rscale[r] + badd;
            }
        }
    }
}

// ---------------- gather + L2 normalize (wave per row); writes f32 and f16 ----------------
__global__ void k_gather_norm(const float* __restrict__ s1, const float* __restrict__ s2,
                              float alpha, float scale, const int* __restrict__ idx,
                              float* __restrict__ outf, _Float16* __restrict__ outh, int nrows) {
    int lane = threadIdx.x & 31;
    int wid  = blockIdx.x * (blockDim.x >> 5) + (threadIdx.x >> 5);
    if (wid >= nrows) return;
    long r = idx[wid];
    const float* p1 = s1 + r * D_DIM;
    float v0 = p1[lane], v1 = p1[32 + lane];
    if (s2) {
        const float* p2 = s2 + r * D_DIM;
        v0 += alpha * p2[lane];
        v1 += alpha * p2[32 + lane];
    }
    v0 *= scale; v1 *= scale;
    float ss = v0 * v0 + v1 * v1;
#pragma unroll
    for (int msk = 1; msk < 32; msk <<= 1) ss += __shfl_xor(ss, msk, 32);
    float inv = 1.0f / (sqrtf(ss) + 1e-8f);
    v0 *= inv; v1 *= inv;
    long o = (long)wid * D_DIM;
    outf[o + lane] = v0;       outf[o + 32 + lane] = v1;
    outh[o + lane] = (_Float16)v0;
    outh[o + 32 + lane] = (_Float16)v1;
}

// ---------------- positive-pair dots: pos[r] = dot(a_n[r], b_n[r]) / t ----------------
__global__ void k_pos(const float* __restrict__ a, const float* __restrict__ b,
                      float* __restrict__ pos, int nrows) {
    int lane = threadIdx.x & 31;
    int wid  = blockIdx.x * (blockDim.x >> 5) + (threadIdx.x >> 5);
    if (wid >= nrows) return;
    long o = (long)wid * D_DIM;
    float d = a[o + lane] * b[o + lane] + a[o + 32 + lane] * b[o + 32 + lane];
#pragma unroll
    for (int msk = 1; msk < 32; msk <<= 1) d += __shfl_xor(d, msk, 32);
    if (lane == 0) pos[wid] = d * (1.0f / TEMP_T);
}

// ---------------- fused logits + logsumexp (WMMA streaming) ----------------
// Inputs are unit vectors => |logit| <= 1/TEMP = 5, exp() can never overflow f32,
// so logsumexp needs no max tracking: per-lane partial sums, one reduction at the end.
__global__ void k_logits_lse(const _Float16* __restrict__ ah, const _Float16* __restrict__ bh,
                             float* __restrict__ lse) {
    int lane = threadIdx.x & 31;
    int wid  = blockIdx.x * (blockDim.x >> 5) + (threadIdx.x >> 5);
    if (wid >= B_BATCH / 16) return;
    int rowBase = wid * 16;

    int off8  = (lane < 16) ? 0 : 8;
    int off16 = (lane < 16) ? 0 : 16;
    const _Float16* ap = ah + (long)(rowBase + (lane & 15)) * D_DIM;

    v8h a00 = *(const v8h*)(ap + off16);
    v8h a01 = *(const v8h*)(ap + 8 + off16);
    v8h a10 = *(const v8h*)(ap + 32 + off16);
    v8h a11 = *(const v8h*)(ap + 40 + off16);
    v16h a0, a1;
#pragma unroll
    for (int i = 0; i < 8; ++i) {
        a0[i] = a00[i]; a0[8 + i] = a01[i];
        a1[i] = a10[i]; a1[8 + i] = a11[i];
    }

    float srun[8];
#pragma unroll
    for (int r = 0; r < 8; ++r) srun[r] = 0.0f;

    const float inv_t = 1.0f / TEMP_T;
    for (int ct = 0; ct < B_BATCH / 16; ++ct) {
        const _Float16* bp = bh + (long)(ct * 16 + (lane & 15)) * D_DIM + off16;
        __builtin_prefetch(bp + 16 * D_DIM, 0, 0);   // next column tile
        v16h b0 = *(const v16h*)bp;
        v16h b1 = *(const v16h*)(bp + 32);
        v8f c = {};
        c = __builtin_amdgcn_wmma_f32_16x16x32_f16(false, a0, false, b0, (short)0, c, false, false);
        c = __builtin_amdgcn_wmma_f32_16x16x32_f16(false, a1, false, b1, (short)0, c, false, false);
#pragma unroll
        for (int r = 0; r < 8; ++r) srun[r] += __expf(c[r] * inv_t);
    }

#pragma unroll
    for (int r = 0; r < 8; ++r) {
        float s = srun[r];
#pragma unroll
        for (int msk = 1; msk < 16; msk <<= 1) s += __shfl_xor(s, msk, 32);
        srun[r] = s;
    }
    if ((lane & 15) == 0) {
#pragma unroll
        for (int r = 0; r < 8; ++r)
            lse[rowBase + r + off8] = __logf(srun[r]);
    }
}

// ---------------- NCE finalize: out[slot] += coeff * mean(lse - pos) ----------------
__global__ void k_nce_finalize(const float* __restrict__ lse, const float* __restrict__ pos,
                               float coeff, float* __restrict__ out, int slot) {
    int i = blockIdx.x * blockDim.x + threadIdx.x;
    float v = (i < B_BATCH) ? (lse[i] - pos[i]) : 0.0f;
#pragma unroll
    for (int msk = 1; msk < 32; msk <<= 1) v += __shfl_xor(v, msk, 32);
    if ((threadIdx.x & 31) == 0) atomicAdd(&out[slot], coeff * v / (float)B_BATCH);
}

// ---------------- BPR loss (wave per sample) ----------------
__global__ void k_bpr(const float* __restrict__ acc, const int* __restrict__ user,
                      const int* __restrict__ posi, const int* __restrict__ negi,
                      float* __restrict__ out) {
    int lane = threadIdx.x & 31;
    int wid  = blockIdx.x * (blockDim.x >> 5) + (threadIdx.x >> 5);
    if (wid >= B_BATCH) return;
    const float s = 1.0f / (float)L_LAY;
    const float* ue = acc + (long)user[wid] * D_DIM;
    const float* pe = acc + (long)(U_N + posi[wid]) * D_DIM;
    const float* ne = acc + (long)(U_N + negi[wid]) * D_DIM;
    float u0 = ue[lane] * s, u1 = ue[32 + lane] * s;
    float d = u0 * (ne[lane] - pe[lane]) * s + u1 * (ne[32 + lane] - pe[32 + lane]) * s;
#pragma unroll
    for (int msk = 1; msk < 32; msk <<= 1) d += __shfl_xor(d, msk, 32);
    if (lane == 0) {
        float sp = fmaxf(d, 0.0f) + log1pf(__expf(-fabsf(d)));  // stable softplus
        atomicAdd(&out[0], sp / (float)B_BATCH);
    }
}

// ---------------- reg loss ----------------
__global__ void k_reg(const float* __restrict__ ut, const float* __restrict__ it,
                      const int* __restrict__ user, const int* __restrict__ posi,
                      const int* __restrict__ negi, float* __restrict__ out) {
    int lane = threadIdx.x & 31;
    int wid  = blockIdx.x * (blockDim.x >> 5) + (threadIdx.x >> 5);
    if (wid >= B_BATCH) return;
    const float* ue = ut + (long)user[wid] * D_DIM;
    const float* pe = it + (long)posi[wid] * D_DIM;
    const float* ne = it + (long)negi[wid] * D_DIM;
    float ss = ue[lane] * ue[lane] + ue[32 + lane] * ue[32 + lane]
             + pe[lane] * pe[lane] + pe[32 + lane] * pe[32 + lane]
             + ne[lane] * ne[lane] + ne[32 + lane] * ne[32 + lane];
#pragma unroll
    for (int msk = 1; msk < 32; msk <<= 1) ss += __shfl_xor(ss, msk, 32);
    if (lane == 0) atomicAdd(&out[1], REG_W * 0.5f * ss / (float)B_BATCH);
}

// ======================================================================
extern "C" void kernel_launch(void* const* d_in, const int* in_sizes, int n_in,
                              void* d_out, int out_size, void* d_ws, size_t ws_size,
                              hipStream_t stream) {
    const float* user_table = (const float*)d_in[0];
    const float* item_table = (const float*)d_in[1];
    const float* graph_vals = (const float*)d_in[2];
    const float* W_[4]  = {(const float*)d_in[3],  (const float*)d_in[7],
                           (const float*)d_in[11], (const float*)d_in[15]};
    const float* b_[4]  = {(const float*)d_in[4],  (const float*)d_in[8],
                           (const float*)d_in[12], (const float*)d_in[16]};
    const float* rW_[4] = {(const float*)d_in[5],  (const float*)d_in[9],
                           (const float*)d_in[13], (const float*)d_in[17]};
    const float* rb_[4] = {(const float*)d_in[6],  (const float*)d_in[10],
                           (const float*)d_in[14], (const float*)d_in[18]};
    const int* graph_rows = (const int*)d_in[19];
    const int* graph_cols = (const int*)d_in[20];
    const int* src_[4] = {(const int*)d_in[21], (const int*)d_in[23],
                          (const int*)d_in[25], (const int*)d_in[27]};
    const int* dst_[4] = {(const int*)d_in[22], (const int*)d_in[24],
                          (const int*)d_in[26], (const int*)d_in[28]};
    const int* userI = (const int*)d_in[29];
    const int* posI  = (const int*)d_in[30];
    const int* negI  = (const int*)d_in[31];
    float* out = (float*)d_out;

    // ---- workspace layout ----
    const long N64 = (long)N_NODES * D_DIM;
    const long U64 = (long)U_N * D_DIM;
    const long I64 = (long)I_N * D_DIM;
    const long B64 = (long)B_BATCH * D_DIM;
    float* w    = (float*)d_ws;
    float* embA = w;            w += N64;
    float* embB = w;            w += N64;
    float* acc  = w;            w += N64;
    float* hu0  = w;            w += U64;
    float* hu1  = w;            w += U64;
    float* hi0  = w;            w += I64;
    float* hi1  = w;            w += I64;
    float* degO = w;            w += U_N;
    float* degI = w;            w += U_N;
    float* a_n  = w;            w += B64;
    float* b_n  = w;            w += B64;
    float* posb = w;            w += B_BATCH;
    float* lseb = w;            w += B_BATCH;
    _Float16* a_h = (_Float16*)w;   w += B64 / 2;
    _Float16* b_h = (_Float16*)w;   w += B64 / 2;
    _Float16* packW = (_Float16*)w; w += 8192 / 2;   // 8192 halves: [W | rW] packed
    float* feat = embA;   // reuse after propagation is done
    float* agg  = embB;

    auto NB = [](long n, int bs) { return (unsigned)((n + bs - 1) / bs); };

    k_zero<<<1, 32, 0, stream>>>(out, 4);

    // ---------- graph propagation: 3 layers + accumulate ----------
    k_concat<<<NB(N64, 256), 256, 0, stream>>>(user_table, item_table, embA);
    k_zero<<<NB(N64, 256), 256, 0, stream>>>(acc, N64);
    const int NNZ = 2 * E_EDGES;
    k_zero<<<NB(N64, 256), 256, 0, stream>>>(embB, N64);
    k_scatter<<<NB((long)NNZ * 32, 256), 256, 0, stream>>>(graph_rows, graph_cols, graph_vals, embA, embB, NNZ);
    k_acc<<<NB(N64, 256), 256, 0, stream>>>(acc, embB, N64);
    k_zero<<<NB(N64, 256), 256, 0, stream>>>(embA, N64);
    k_scatter<<<NB((long)NNZ * 32, 256), 256, 0, stream>>>(graph_rows, graph_cols, graph_vals, embB, embA, NNZ);
    k_acc<<<NB(N64, 256), 256, 0, stream>>>(acc, embA, N64);
    k_zero<<<NB(N64, 256), 256, 0, stream>>>(embB, N64);
    k_scatter<<<NB((long)NNZ * 32, 256), 256, 0, stream>>>(graph_rows, graph_cols, graph_vals, embA, embB, NNZ);
    k_acc<<<NB(N64, 256), 256, 0, stream>>>(acc, embB, N64);
    // final = acc / 3 (scale folded into consumers)

    // ---------- 4 light encoders (packed-W WMMA GEMMs + scatter) ----------
    const float* X_[4]  = {user_table, user_table, item_table, item_table};
    const int    n_[4]  = {U_N, U_N, I_N, I_N};
    const int    nE_[4] = {EU_EDGES, EU_EDGES, EI_EDGES, EI_EDGES};
    float*       h_[4]  = {hu0, hu1, hi0, hi1};
    for (int enc = 0; enc < 4; ++enc) {
        int n  = n_[enc];
        int nE = nE_[enc];
        k_packW<<<32, 256, 0, stream>>>(W_[enc], rW_[enc], packW);
        k_zero<<<NB(n, 256), 256, 0, stream>>>(degO, n);
        k_zero<<<NB(n, 256), 256, 0, stream>>>(degI, n);
        k_deg<<<NB(nE, 256), 256, 0, stream>>>(src_[enc], dst_[enc], degO, degI, nE);
        long nwaves = n / 16;
        // feat = (X @ W) * deg_out^-1/2
        k_encoder_gemm<<<NB(nwaves * 32, 128), 128, 0, stream>>>(
            X_[enc], packW, nullptr, nullptr, degO, nullptr, feat, n, 0);
        // agg[dst] += feat[src]
        k_zero<<<NB((long)n * D_DIM, 256), 256, 0, stream>>>(agg, (long)n * D_DIM);
        k_scatter<<<NB((long)nE * 32, 256), 256, 0, stream>>>(dst_[enc], src_[enc], nullptr, feat, agg, nE);
        // h = X @ rW + agg * deg_in^-1/2 + b + rb
        k_encoder_gemm<<<NB(nwaves * 32, 128), 128, 0, stream>>>(
            X_[enc], packW + 4096, b_[enc], rb_[enc], degI, agg, h_[enc], n, 1);
    }

    // ---------- InfoNCE (fused WMMA logits + logsumexp) ----------
    auto run_nce = [&](const float* s1a, const float* s2a, float aalpha, float ascale, const int* ia,
                       const float* s1b, const float* s2b, float balpha, float bscale, const int* ib,
                       float coeff, int slot) {
        k_gather_norm<<<NB((long)B_BATCH * 32, 256), 256, 0, stream>>>(s1a, s2a, aalpha, ascale, ia, a_n, a_h, B_BATCH);
        k_gather_norm<<<NB((long)B_BATCH * 32, 256), 256, 0, stream>>>(s1b, s2b, balpha, bscale, ib, b_n, b_h, B_BATCH);
        k_pos<<<NB((long)B_BATCH * 32, 256), 256, 0, stream>>>(a_n, b_n, posb, B_BATCH);
        k_logits_lse<<<NB((long)(B_BATCH / 16) * 32, 128), 128, 0, stream>>>(a_h, b_h, lseb);
        k_nce_finalize<<<B_BATCH / 256, 256, 0, stream>>>(lseb, posb, coeff, out, slot);
    };

    const float third = 1.0f / (float)L_LAY;
    run_nce(hu0, nullptr, 0.0f, 1.0f, userI, hu1, nullptr, 0.0f, 1.0f, userI, INTRA_W, 3);
    run_nce(hi0, nullptr, 0.0f, 1.0f, posI,  hi1, nullptr, 0.0f, 1.0f, posI,  INTRA_W, 3);
    run_nce(acc, nullptr, 0.0f, third, userI, hu0, hu1, 1.0f, 0.5f, userI, SSL_W, 2);
    run_nce(acc + (long)U_N * D_DIM, nullptr, 0.0f, third, posI, hi0, hi1, 1.0f, 0.5f, posI, SSL_W, 2);

    // ---------- BPR + reg ----------
    k_bpr<<<NB((long)B_BATCH * 32, 256), 256, 0, stream>>>(acc, userI, posI, negI, out);
    k_reg<<<NB((long)B_BATCH * 32, 256), 256, 0, stream>>>(user_table, item_table, userI, posI, negI, out);
}
